// MultiHeadAttention_10350871183565
// MI455X (gfx1250) — matrix operand order repaired and checked by reading, not compiled, not run on hardware
//
#include <hip/hip_runtime.h>
#include <hip/hip_bf16.h>

typedef __attribute__((ext_vector_type(16))) _Float16 v16h;
typedef __attribute__((ext_vector_type(8)))  _Float16 v8h;
typedef __attribute__((ext_vector_type(8)))  float    v8f;

#define EMBED 1024
#define HEADS 16
#define HDIM  64
#define NB    4
#define SEQ   2048
#define ROWS  (NB * SEQ)   // 8192

#define WMMA_F16(a, b, c) \
  __builtin_amdgcn_wmma_f32_16x16x32_f16(false, (a), false, (b), (short)0, (c), false, false)

static __device__ __forceinline__ int lane_id() { return (int)(threadIdx.x & 31u); }

// ---------------- WMMA fragment loaders (wave32, 16-bit ops) ----------------
// A fragment: 16(M) x 32(K) tile from row-major f16, K contiguous.
// Lane L holds row M=L&15; lanes 0-15 take K = k0+{0..7, 16..23}, lanes 16-31 take +8.
static __device__ __forceinline__ v16h load_a_f16(const _Float16* base, int ld, int k0) {
  int lane = lane_id();
  const _Float16* p = base + (size_t)(lane & 15) * ld + k0 + ((lane & 16) ? 8 : 0);
  v8h lo = *(const v8h*)p;
  v8h hi = *(const v8h*)(p + 16);
  v16h f;
#pragma unroll
  for (int i = 0; i < 8; ++i) { f[i] = lo[i]; f[i + 8] = hi[i]; }
  return f;
}

// B fragment: 32(K) x 16(N) tile; source is B^T row-major (row n = output column,
// K contiguous). Lane L holds column N=L&15; lanes 0-15 take K = k0+{0..7, 16..23}, lanes 16-31 take +8.
static __device__ __forceinline__ v16h load_b_f16(const _Float16* baseT, int ld, int k0) {
  int lane = lane_id();
  const _Float16* p = baseT + (size_t)(lane & 15) * ld + k0 + ((lane & 16) ? 8 : 0);
  v8h lo = *(const v8h*)p;
  v8h hi = *(const v8h*)(p + 16);
  v16h f;
#pragma unroll
  for (int i = 0; i < 8; ++i) { f[i] = lo[i]; f[i + 8] = hi[i]; }
  return f;
}

// ---------------- Kernel 0: bulk f32 -> f16 convert (one pass over inputs/weights) ------
__global__ __launch_bounds__(256)
void cvt_f16_kernel(const float* __restrict__ src, _Float16* __restrict__ dst, int n) {
  int i = ((int)blockIdx.x * 256 + (int)threadIdx.x) * 8;
  if (i + 8 <= n) {
    float4 a = *(const float4*)(src + i);
    float4 b = *(const float4*)(src + i + 4);
    v8h h;
    h[0] = (_Float16)a.x; h[1] = (_Float16)a.y; h[2] = (_Float16)a.z; h[3] = (_Float16)a.w;
    h[4] = (_Float16)b.x; h[5] = (_Float16)b.y; h[6] = (_Float16)b.z; h[7] = (_Float16)b.w;
    *(v8h*)(dst + i) = h;
  }
}

// ---------------- Kernel 1: input projections Xh @ Wh^T -> f16, head-split --------------
// Per wave: 32(M) x 64(N) tile strip (2 M-tiles x 4 N-tiles, 8 accumulators).
// store_mode 0: out[((n*H+h)*SEQ + s)*64 + d]        (Q, K)
// store_mode 1: out[((n*H+h)*64 + d)*SEQ + s]        (V transposed for the PV B-operand)
__global__ __launch_bounds__(128)
void proj_kernel(const _Float16* __restrict__ X, const _Float16* __restrict__ W,
                 _Float16* __restrict__ out, int store_mode) {
  int wid = (int)blockIdx.x * 4 + (int)(threadIdx.x >> 5);
  const int colStrips = EMBED / 64;                // 16
  int i0 = (wid / colStrips) * 32;                 // 32 rows
  int j0 = (wid % colStrips) * 64;                 // 64 cols
  if (i0 >= ROWS) return;

  v8f zero = {};
  v8f acc[2][4] = {{zero, zero, zero, zero}, {zero, zero, zero, zero}};
  for (int k0 = 0; k0 < EMBED; k0 += 32) {
    v16h a0 = load_a_f16(X + (size_t)i0 * EMBED, EMBED, k0);
    v16h a1 = load_a_f16(X + (size_t)(i0 + 16) * EMBED, EMBED, k0);
#pragma unroll
    for (int t = 0; t < 4; ++t) {
      v16h b = load_b_f16(W + (size_t)(j0 + t * 16) * EMBED, EMBED, k0);
      acc[0][t] = WMMA_F16(a0, b, acc[0][t]);
      acc[1][t] = WMMA_F16(a1, b, acc[1][t]);
    }
  }
  int lane = lane_id();
  int ln = lane & 15;
  int mh = (lane & 16) ? 8 : 0;
#pragma unroll
  for (int r = 0; r < 2; ++r) {
#pragma unroll
    for (int t = 0; t < 4; ++t) {
#pragma unroll
      for (int v = 0; v < 8; ++v) {
        int gi = i0 + r * 16 + v + mh;             // n*SEQ + s
        int gj = j0 + t * 16 + ln;                 // h*64 + d
        int n = gi >> 11, s = gi & (SEQ - 1);
        int h = gj >> 6,  d = gj & 63;
        size_t idx = (store_mode == 0)
            ? ((size_t)((n * HEADS + h) * SEQ + s)) * HDIM + d
            : ((size_t)((n * HEADS + h) * HDIM + d)) * SEQ + s;
        out[idx] = (_Float16)acc[r][t][v];
      }
    }
  }
}

// ---------------- Kernel 2: fused flash attention per (n, h, 16-query tile) -------------
__global__ __launch_bounds__(128)
void attn_kernel(const _Float16* __restrict__ Qh, const _Float16* __restrict__ Kh,
                 const _Float16* __restrict__ VhT, const int* __restrict__ mask,
                 _Float16* __restrict__ concat) {
  __shared__ _Float16 smem[4 * 16 * 32];           // 1KB per wave: P block 16x32
  int wslot = (int)(threadIdx.x >> 5);
  _Float16* P = smem + wslot * (16 * 32);

  int wid = (int)blockIdx.x * 4 + wslot;
  const int qTiles = SEQ / 16;                     // 128
  int qt = wid % qTiles;
  int nh = wid / qTiles;                           // 0..63
  int h = nh % HEADS, n = nh / HEADS;
  int qb = qt * 16;

  const _Float16* qhead  = Qh  + (size_t)nh * SEQ * HDIM;
  const _Float16* khead  = Kh  + (size_t)nh * SEQ * HDIM;
  const _Float16* vheadT = VhT + (size_t)nh * HDIM * SEQ;
  const int* mrow = mask + (size_t)n * SEQ * SEQ;

  int lane = lane_id();
  int ln = lane & 15;
  int mh = (lane & 16) ? 8 : 0;

  // Q tile as A operand (K-dim = 64 -> two 16x32 fragments), loaded once.
  v16h aq0 = load_a_f16(qhead + (size_t)qb * HDIM, HDIM, 0);
  v16h aq1 = load_a_f16(qhead + (size_t)qb * HDIM, HDIM, 32);

  v8f zero = {};
  v8f o[4] = {zero, zero, zero, zero};             // 16x64 output accumulator
  float m[8], l[8];
#pragma unroll
  for (int v = 0; v < 8; ++v) { m[v] = -1e30f; l[v] = 0.0f; }

  for (int kb = 0; kb < SEQ; kb += 32) {
    if (kb + 32 < SEQ)
      __builtin_prefetch(khead + (size_t)(kb + 32) * HDIM, 0, 0);

    // energy tiles: E = Q @ K^T  (K rows are B columns; d contiguous)
    v16h bk00 = load_b_f16(khead + (size_t)kb * HDIM,        HDIM, 0);
    v16h bk01 = load_b_f16(khead + (size_t)kb * HDIM,        HDIM, 32);
    v16h bk10 = load_b_f16(khead + (size_t)(kb + 16) * HDIM, HDIM, 0);
    v16h bk11 = load_b_f16(khead + (size_t)(kb + 16) * HDIM, HDIM, 32);
    v8f e0 = WMMA_F16(aq0, bk00, zero);
    e0     = WMMA_F16(aq1, bk01, e0);
    v8f e1 = WMMA_F16(aq0, bk10, zero);
    e1     = WMMA_F16(aq1, bk11, e1);

    // mask (before scaling, as in reference), online softmax per row.
    // C/D layout: VGPR v <-> row v (lanes 0-15) / v+8 (lanes 16-31); lane = key col.
#pragma unroll
    for (int v = 0; v < 8; ++v) {
      int row = qb + v + mh;
      float x0 = (mrow[(size_t)row * SEQ + (kb + ln)]      == 0) ? -10000.0f : e0[v];
      float x1 = (mrow[(size_t)row * SEQ + (kb + 16 + ln)] == 0) ? -10000.0f : e1[v];
      x0 *= 0.125f;                                // 1/sqrt(HEAD_DIM)
      x1 *= 0.125f;
      float bm = fmaxf(x0, x1);
#pragma unroll
      for (int off = 1; off < 16; off <<= 1) bm = fmaxf(bm, __shfl_xor(bm, off, 16));
      float mn = fmaxf(m[v], bm);
      float alpha = __expf(m[v] - mn);
      float p0 = __expf(x0 - mn);
      float p1 = __expf(x1 - mn);
      float rs = p0 + p1;
#pragma unroll
      for (int off = 1; off < 16; off <<= 1) rs += __shfl_xor(rs, off, 16);
      l[v] = l[v] * alpha + rs;
      m[v] = mn;
#pragma unroll
      for (int c = 0; c < 4; ++c) o[c][v] *= alpha;
      // stage P (D layout) into LDS so it can be re-read in A layout
      P[(v + mh) * 32 + ln]      = (_Float16)p0;
      P[(v + mh) * 32 + 16 + ln] = (_Float16)p1;
    }
    __syncthreads();                               // order cross-lane LDS store -> load

    // O += P @ V  (V^T rows are B columns; keys contiguous)
    v16h pa = load_a_f16(P, 32, 0);
#pragma unroll
    for (int c = 0; c < 4; ++c) {
      v16h bv = load_b_f16(vheadT + (size_t)(c * 16) * SEQ + kb, SEQ, 0);
      o[c] = WMMA_F16(pa, bv, o[c]);
    }
  }

  // normalize and write concat [N*SEQ, EMBED] f16: col = h*64 + c*16 + ln
#pragma unroll
  for (int c = 0; c < 4; ++c) {
#pragma unroll
    for (int v = 0; v < 8; ++v) {
      int row = qb + v + mh;
      float val = o[c][v] / l[v];
      size_t idx = ((size_t)(n * SEQ + row)) * EMBED + h * HDIM + c * 16 + ln;
      concat[idx] = (_Float16)val;
    }
  }
}

// ---------------- Kernel 3: output projection concat @ Wo^T + bo -> f32 -----------------
__global__ __launch_bounds__(128)
void outproj_kernel(const _Float16* __restrict__ Cc, const _Float16* __restrict__ Wo,
                    const float* __restrict__ bo, float* __restrict__ out) {
  int wid = (int)blockIdx.x * 4 + (int)(threadIdx.x >> 5);
  const int colStrips = EMBED / 64;
  int i0 = (wid / colStrips) * 32;
  int j0 = (wid % colStrips) * 64;
  if (i0 >= ROWS) return;

  v8f zero = {};
  v8f acc[2][4] = {{zero, zero, zero, zero}, {zero, zero, zero, zero}};
  for (int k0 = 0; k0 < EMBED; k0 += 32) {
    v16h a0 = load_a_f16(Cc + (size_t)i0 * EMBED, EMBED, k0);
    v16h a1 = load_a_f16(Cc + (size_t)(i0 + 16) * EMBED, EMBED, k0);
#pragma unroll
    for (int t = 0; t < 4; ++t) {
      v16h b = load_b_f16(Wo + (size_t)(j0 + t * 16) * EMBED, EMBED, k0);
      acc[0][t] = WMMA_F16(a0, b, acc[0][t]);
      acc[1][t] = WMMA_F16(a1, b, acc[1][t]);
    }
  }
  int lane = lane_id();
  int ln = lane & 15;
  int mh = (lane & 16) ? 8 : 0;
#pragma unroll
  for (int r = 0; r < 2; ++r) {
#pragma unroll
    for (int t = 0; t < 4; ++t) {
#pragma unroll
      for (int v = 0; v < 8; ++v) {
        int gi = i0 + r * 16 + v + mh;
        int gj = j0 + t * 16 + ln;
        out[(size_t)gi * EMBED + gj] = acc[r][t][v] + bo[gj];
      }
    }
  }
}

// ---------------- launcher ----------------
extern "C" void kernel_launch(void* const* d_in, const int* in_sizes, int n_in,
                              void* d_out, int out_size, void* d_ws, size_t ws_size,
                              hipStream_t stream) {
  (void)in_sizes; (void)n_in; (void)out_size; (void)ws_size;
  const float* query  = (const float*)d_in[0];
  const float* keys   = (const float*)d_in[1];
  const float* values = (const float*)d_in[2];
  const int*   mask   = (const int*)d_in[3];
  const float* Wq = (const float*)d_in[4];
  const float* Wk = (const float*)d_in[5];
  const float* Wv = (const float*)d_in[6];
  const float* Wo = (const float*)d_in[7];
  const float* bo = (const float*)d_in[8];
  float* out = (float*)d_out;

  const size_t ACT = (size_t)ROWS * EMBED;                 // 8.39M elems
  const size_t WEL = (size_t)EMBED * EMBED;                // 1.05M elems
  _Float16* q_ws  = (_Float16*)d_ws;                       // [N,H,S,64]  16MB
  _Float16* k_ws  = q_ws  + ACT;                           // [N,H,S,64]  16MB
  _Float16* vT_ws = k_ws  + ACT;                           // [N,H,64,S]  16MB
  _Float16* cc_ws = vT_ws + ACT;                           // [N*S,1024]  16MB
  _Float16* xq_h  = cc_ws + ACT;                           // f16 copies of activations
  _Float16* xk_h  = xq_h  + ACT;
  _Float16* xv_h  = xk_h  + ACT;
  _Float16* wq_h  = xv_h  + ACT;                           // f16 copies of weights
  _Float16* wk_h  = wq_h  + WEL;
  _Float16* wv_h  = wk_h  + WEL;
  _Float16* wo_h  = wv_h  + WEL;

  // one-pass f32 -> f16 staging (memory-bound, ~6us total at 23.3 TB/s)
  {
    dim3 cb(256);
    dim3 cgA((unsigned)(ACT / (256 * 8)));
    dim3 cgW((unsigned)(WEL / (256 * 8)));
    cvt_f16_kernel<<<cgA, cb, 0, stream>>>(query,  xq_h, (int)ACT);
    cvt_f16_kernel<<<cgA, cb, 0, stream>>>(keys,   xk_h, (int)ACT);
    cvt_f16_kernel<<<cgA, cb, 0, stream>>>(values, xv_h, (int)ACT);
    cvt_f16_kernel<<<cgW, cb, 0, stream>>>(Wq, wq_h, (int)WEL);
    cvt_f16_kernel<<<cgW, cb, 0, stream>>>(Wk, wk_h, (int)WEL);
    cvt_f16_kernel<<<cgW, cb, 0, stream>>>(Wv, wv_h, (int)WEL);
    cvt_f16_kernel<<<cgW, cb, 0, stream>>>(Wo, wo_h, (int)WEL);
  }

  dim3 blk(128);                                           // 4 waves/block (wave32)
  int wavesProj = (ROWS / 32) * (EMBED / 64);              // 4096
  dim3 grdProj(wavesProj / 4);
  proj_kernel<<<grdProj, blk, 0, stream>>>(xq_h, wq_h, q_ws,  0);
  proj_kernel<<<grdProj, blk, 0, stream>>>(xk_h, wk_h, k_ws,  0);
  proj_kernel<<<grdProj, blk, 0, stream>>>(xv_h, wv_h, vT_ws, 1);

  int wavesAttn = NB * HEADS * (SEQ / 16);                 // 8192
  attn_kernel<<<dim3(wavesAttn / 4), blk, 0, stream>>>(q_ws, k_ws, vT_ws, mask, cc_ws);

  outproj_kernel<<<grdProj, blk, 0, stream>>>(cc_ws, wo_h, bo, out);
}
